// GMM_1451698946616
// MI455X (gfx1250) — matrix operand order repaired and checked by reading, not compiled, not run on hardware
//
#include <hip/hip_runtime.h>
#include <hip/hip_bf16.h>
#include <stdint.h>

#define LOG_2PI 1.8378770664093453f

typedef __attribute__((ext_vector_type(16))) __bf16 v16bf;
typedef __attribute__((ext_vector_type(8)))  __bf16 v8bf;
typedef __attribute__((ext_vector_type(8)))  float  v8f;
typedef __attribute__((ext_vector_type(4)))  int    v4i;

#if __has_builtin(__builtin_amdgcn_global_load_async_to_lds_b128) && __has_builtin(__builtin_amdgcn_s_wait_asynccnt)
#define USE_ASYNC 1
#else
#define USE_ASYNC 0
#endif

#define F_DIM 1024
#define K_BPE 256
#define L_G   8
#define KL    2048
#define M_TOT 32768   // B*T = 16*2048
#define TSTR  40      // LDS tile row stride in elements (80 B: bank-conflict-free)

// DPP cross-lane move: quad_perm / row ops, full-rate VALU (no LDS traffic).
#define DPPF(x, ctrl) \
  __int_as_float(__builtin_amdgcn_update_dpp(__float_as_int(x), __float_as_int(x), (ctrl), 0xF, 0xF, true))
#define DPP_XOR1 0xB1   // quad_perm [1,0,3,2]
#define DPP_XOR2 0x4E   // quad_perm [2,3,0,1]
#define DPP_HMIR 0x141  // row_half_mirror (swap 4-groups within each 8)

// ---------------------------------------------------------------------------
// Prep 1: per (k,l) row build bf16 weights W2 = -0.5*exp(-log_cov),
// W1 = mu*exp(-log_cov), and bias0 = -0.5*D*log2pi - 0.5*sum(log_cov)
//                                    - 0.5*sum(mu^2*inv_var)
// ---------------------------------------------------------------------------
__global__ __launch_bounds__(256) void prep_w(const float* __restrict__ mu,
                                              const float* __restrict__ log_cov,
                                              __bf16* __restrict__ w2,
                                              __bf16* __restrict__ w1,
                                              float*  __restrict__ bias) {
  const int kl  = blockIdx.x;
  const int tid = threadIdx.x;
  const size_t row = (size_t)kl * F_DIM;
  float s_lc = 0.f, s_m2 = 0.f;
#pragma unroll
  for (int j = 0; j < F_DIM / 256; ++j) {
    int f = j * 256 + tid;
    float lc = log_cov[row + f];
    float m  = mu[row + f];
    float iv = __expf(-lc);
    w2[row + f] = (__bf16)(-0.5f * iv);
    w1[row + f] = (__bf16)(m * iv);
    s_lc += lc;
    s_m2 += m * m * iv;
  }
  __shared__ float r1[256], r2[256];
  r1[tid] = s_lc; r2[tid] = s_m2;
  __syncthreads();
  for (int s = 128; s > 0; s >>= 1) {
    if (tid < s) { r1[tid] += r1[tid + s]; r2[tid] += r2[tid + s]; }
    __syncthreads();
  }
  if (tid == 0)
    bias[kl] = -0.5f * ((float)F_DIM * LOG_2PI) - 0.5f * r1[0] - 0.5f * r2[0];
}

// Prep 2: fold log_softmax(log_pi, axis=L) into bias. One thread per k.
__global__ __launch_bounds__(256) void prep_bias(const float* __restrict__ log_pi,
                                                 float* __restrict__ bias) {
  int k = threadIdx.x;
  float lp[L_G];
  float mx = -3.0e38f;
#pragma unroll
  for (int l = 0; l < L_G; ++l) { lp[l] = log_pi[k * L_G + l]; mx = fmaxf(mx, lp[l]); }
  float s = 0.f;
#pragma unroll
  for (int l = 0; l < L_G; ++l) s += __expf(lp[l] - mx);
  float lse = mx + __logf(s);
#pragma unroll
  for (int l = 0; l < L_G; ++l) bias[k * L_G + l] += lp[l] - lse;
}

// 32-byte weight-row chunk: global bf16 -> LDS (async on CDNA5).
__device__ __forceinline__ void copy_w16(const __bf16* g, __bf16* l) {
#if USE_ASYNC
  typedef __attribute__((address_space(1))) v4i* gp4;
  typedef __attribute__((address_space(3))) v4i* lp4;
  v4i* gv = (v4i*)g;
  v4i* lv = (v4i*)l;
  __builtin_amdgcn_global_load_async_to_lds_b128((gp4)gv,       (lp4)lv,       0, 0);
  __builtin_amdgcn_global_load_async_to_lds_b128((gp4)(gv + 1), (lp4)(lv + 1), 0, 0);
#else
  *(v8bf*)l             = *(const v8bf*)g;
  *((v8bf*)l + 1)       = *((const v8bf*)g + 1);
#endif
}

// ---------------------------------------------------------------------------
// Main fused GEMM + logsumexp kernel, double-buffered software pipeline.
// Block tile: 128 (M=bt) x 128 (N=kl); K loop over f in steps of 32;
// A = [X^2 ; X] built on the fly. 8 wave32s: 4(M) x 2(N); each wave 2x4
// accumulators of 16x16 -> 16 v_wmma_f32_16x16x32_bf16 per K-step.
// ---------------------------------------------------------------------------
__global__ __launch_bounds__(256) void gmm_gemm(const float*  __restrict__ X,
                                                const __bf16* __restrict__ w2g,
                                                const __bf16* __restrict__ w1g,
                                                const float*  __restrict__ bias,
                                                float* __restrict__ out) {
  __shared__ __align__(32) __bf16 A2t[2][128 * TSTR];
  __shared__ __align__(32) __bf16 A1t[2][128 * TSTR];
  __shared__ __align__(32) __bf16 B2t[2][128 * TSTR];
  __shared__ __align__(32) __bf16 B1t[2][128 * TSTR];

  const int tid  = threadIdx.x;
  const int lane = tid & 31;
  const int wave = tid >> 5;
  const int wm   = wave & 3;          // M sub-block (32 rows each)
  const int wn   = wave >> 2;         // N sub-block (64 cols each)
  const int Nbase = blockIdx.x * 128;
  const int Mbase = blockIdx.y * 128;

  v8f acc[2][4] = {};

  // copy-role mapping: each thread services one 16-elem half of one tile row
  const int r    = tid >> 1;
  const int c16  = (tid & 1) * 16;
  const int ldso = r * TSTR + c16;
  const float*  xrow  = X   + (size_t)(Mbase + r) * F_DIM + c16;
  const __bf16* w2row = w2g + (size_t)(Nbase + r) * F_DIM + c16;
  const __bf16* w1row = w1g + (size_t)(Nbase + r) * F_DIM + c16;

  union V16 { v16bf v; v8bf h[2]; };

  // ---- prologue: fill buffer 0 (fb = 0) ----
  {
    float xv[16];
    const float4* xp4 = (const float4*)xrow;
#pragma unroll
    for (int q = 0; q < 4; ++q) {
      float4 t = xp4[q];
      xv[q * 4 + 0] = t.x; xv[q * 4 + 1] = t.y;
      xv[q * 4 + 2] = t.z; xv[q * 4 + 3] = t.w;
    }
    V16 a1v, a2v;
#pragma unroll
    for (int e = 0; e < 16; ++e) {
      a1v.v[e] = (__bf16)xv[e];
      a2v.v[e] = (__bf16)(xv[e] * xv[e]);
    }
    *(v8bf*)&A1t[0][ldso]     = a1v.h[0];
    *(v8bf*)&A1t[0][ldso + 8] = a1v.h[1];
    *(v8bf*)&A2t[0][ldso]     = a2v.h[0];
    *(v8bf*)&A2t[0][ldso + 8] = a2v.h[1];
    copy_w16(w2row, &B2t[0][ldso]);
    copy_w16(w1row, &B1t[0][ldso]);
#if USE_ASYNC
    __builtin_amdgcn_s_wait_asynccnt(0);
#endif
  }
  __syncthreads();

  const int lrow = lane & 15;
  const int ak   = (lane >> 4) * 8;    // A: interleaved 8-elem K chunks
  const int bk   = (lane >> 4) * 8;    // B: interleaved 8-elem K chunks

  for (int it = 0; it < F_DIM / 32; ++it) {
    const int  cur  = it & 1;
    const int  nxt  = cur ^ 1;
    const int  fb2  = (it + 1) * 32;
    const bool more = fb2 < F_DIM;

    // ---- issue next tile's global traffic before computing ----
    float xv[16];
    if (more) {
      const float4* xp4 = (const float4*)(xrow + fb2);
#pragma unroll
      for (int q = 0; q < 4; ++q) {
        float4 t = xp4[q];
        xv[q * 4 + 0] = t.x; xv[q * 4 + 1] = t.y;
        xv[q * 4 + 2] = t.z; xv[q * 4 + 3] = t.w;
      }
      copy_w16(w2row + fb2, &B2t[nxt][ldso]);
      copy_w16(w1row + fb2, &B1t[nxt][ldso]);
    }

    // ---- fragments from current buffer ----
    // A (16-bit 16x32 layout): lane<16 holds K[0..7]+K[16..23], lane>=16
    // holds K[8..15]+K[24..31] of row M=lane%16.
    V16 a2f[2], a1f[2];
#pragma unroll
    for (int mi = 0; mi < 2; ++mi) {
      int base = (wm * 32 + mi * 16 + lrow) * TSTR;
      a2f[mi].h[0] = *(const v8bf*)&A2t[cur][base + ak];
      a2f[mi].h[1] = *(const v8bf*)&A2t[cur][base + 16 + ak];
      a1f[mi].h[0] = *(const v8bf*)&A1t[cur][base + ak];
      a1f[mi].h[1] = *(const v8bf*)&A1t[cur][base + 16 + ak];
    }
    // B (32x16): lane%16 = N column, lane<16 K[0..7]+K[16..23], lane>=16 K[8..15]+K[24..31].
    V16 b2f[4], b1f[4];
#pragma unroll
    for (int ni = 0; ni < 4; ++ni) {
      int base = (wn * 64 + ni * 16 + lrow) * TSTR + bk;
      b2f[ni].h[0] = *(const v8bf*)&B2t[cur][base];
      b2f[ni].h[1] = *(const v8bf*)&B2t[cur][base + 16];
      b1f[ni].h[0] = *(const v8bf*)&B1t[cur][base];
      b1f[ni].h[1] = *(const v8bf*)&B1t[cur][base + 16];
    }

#pragma unroll
    for (int mi = 0; mi < 2; ++mi)
#pragma unroll
      for (int ni = 0; ni < 4; ++ni) {
        acc[mi][ni] = __builtin_amdgcn_wmma_f32_16x16x32_bf16(
            false, a2f[mi].v, false, b2f[ni].v, (short)0, acc[mi][ni], false, false);
        acc[mi][ni] = __builtin_amdgcn_wmma_f32_16x16x32_bf16(
            false, a1f[mi].v, false, b1f[ni].v, (short)0, acc[mi][ni], false, false);
      }

    // ---- convert + stage next A tiles while WMMAs drain ----
    if (more) {
      V16 a1v, a2v;
#pragma unroll
      for (int e = 0; e < 16; ++e) {
        a1v.v[e] = (__bf16)xv[e];
        a2v.v[e] = (__bf16)(xv[e] * xv[e]);
      }
      *(v8bf*)&A1t[nxt][ldso]     = a1v.h[0];
      *(v8bf*)&A1t[nxt][ldso + 8] = a1v.h[1];
      *(v8bf*)&A2t[nxt][ldso]     = a2v.h[0];
      *(v8bf*)&A2t[nxt][ldso + 8] = a2v.h[1];
#if USE_ASYNC
      __builtin_amdgcn_s_wait_asynccnt(0);
#endif
    }
    __syncthreads();
  }

  // ---- Epilogue: + bias, logsumexp over L=8 (8 adjacent N lanes) via DPP ----
#pragma unroll
  for (int mi = 0; mi < 2; ++mi)
#pragma unroll
    for (int ni = 0; ni < 4; ++ni) {
      int nglb = Nbase + wn * 64 + ni * 16 + lrow;
      float bv = bias[nglb];
#pragma unroll
      for (int v = 0; v < 8; ++v) {
        float x  = acc[mi][ni][v] + bv;
        float mx = x;
        mx = fmaxf(mx, DPPF(mx, DPP_XOR1));
        mx = fmaxf(mx, DPPF(mx, DPP_XOR2));
        mx = fmaxf(mx, DPPF(mx, DPP_HMIR));
        float e = __expf(x - mx);
        e += DPPF(e, DPP_XOR1);
        e += DPPF(e, DPP_XOR2);
        e += DPPF(e, DPP_HMIR);
        if ((lane & 7) == 0) {
          int m = Mbase + wm * 32 + mi * 16 + v + ((lane >> 4) << 3);
          out[(size_t)m * K_BPE + (nglb >> 3)] = mx + __logf(e);
        }
      }
    }
}

// ---------------------------------------------------------------------------
extern "C" void kernel_launch(void* const* d_in, const int* in_sizes, int n_in,
                              void* d_out, int out_size, void* d_ws, size_t ws_size,
                              hipStream_t stream) {
  const float* X   = (const float*)d_in[0];   // [16,2048,1024]
  const float* mu  = (const float*)d_in[1];   // [256,8,1024]
  const float* lcv = (const float*)d_in[2];   // [256,8,1024]
  const float* lpi = (const float*)d_in[3];   // [256,8]
  float* out = (float*)d_out;                 // [16,2048,256]

  char* ws = (char*)d_ws;
  __bf16* w2   = (__bf16*)ws;                                  // 4 MiB
  __bf16* w1   = (__bf16*)(ws + (size_t)KL * F_DIM * 2);       // 4 MiB
  float*  bias = (float*) (ws + (size_t)KL * F_DIM * 4);       // 8 KiB

  prep_w<<<dim3(KL), dim3(256), 0, stream>>>(mu, lcv, w2, w1, bias);
  prep_bias<<<dim3(1), dim3(256), 0, stream>>>(lpi, bias);
  gmm_gemm<<<dim3(KL / 128, M_TOT / 128), dim3(256), 0, stream>>>(X, w2, w1, bias, out);
}
